// RWKV7FixedEncoder_38190849196278
// MI455X (gfx1250) — matrix operand order repaired and checked by reading, not compiled, not run on hardware
//
#include <hip/hip_runtime.h>
#include <hip/hip_bf16.h>
#include <math.h>

// ---------------------------------------------------------------------------
// RWKV-7 forward for MI455X (gfx1250, wave32, WMMA).
// All GEMM operands are bf16-native in memory (producers emit bf16, weights
// converted+transposed once per layer), so GEMM staging is pure b128 traffic:
// global_load_b128 -> ds_store_b128 -> ds_load_b128 -> v_wmma_f32_16x16x32_bf16.
// Matrix-state recurrence -> register-resident S (64 VGPRs/thread).
// ---------------------------------------------------------------------------

typedef __attribute__((ext_vector_type(16))) __bf16 v16bf;
typedef __attribute__((ext_vector_type(8)))  float  v8f;

#define Cdim   768
#define Hh     12
#define Nn     64
#define Tt     1024
#define Bb     2
#define Mtok   (Bb * Tt)       // 2048 tokens
#define FFNDIM 3072

enum { EPI_NONE = 0, EPI_SIGMOID = 1, EPI_TANH = 2, EPI_RELUSQ = 3 };

// ---------------------------------------------------------------------------
// LayerNorm over C=768. One token per block, 256 threads x 3 channels.
// ---------------------------------------------------------------------------
__global__ void ln_kernel(const float* __restrict__ x, const float* __restrict__ w,
                          const float* __restrict__ b, float* __restrict__ out) {
  int m = blockIdx.x;
  int tid = threadIdx.x;
  __shared__ float ssum, ssq;
  if (tid == 0) { ssum = 0.f; ssq = 0.f; }
  __syncthreads();
  float v0[3];
  float ps = 0.f, pq = 0.f;
#pragma unroll
  for (int i = 0; i < 3; ++i) {
    float t = x[(size_t)m * Cdim + tid * 3 + i];
    v0[i] = t; ps += t; pq += t * t;
  }
  atomicAdd(&ssum, ps);
  atomicAdd(&ssq, pq);
  __syncthreads();
  float mu  = ssum * (1.f / Cdim);
  float var = ssq * (1.f / Cdim) - mu * mu;
  float inv = rsqrtf(var + 1e-5f);
#pragma unroll
  for (int i = 0; i < 3; ++i) {
    int c = tid * 3 + i;
    out[(size_t)m * Cdim + c] = (v0[i] - mu) * inv * w[c] + b[c];
  }
}

// ---------------------------------------------------------------------------
// Weight convert + transpose: Wt[n*K + k] = (bf16) W[k*N + n].  Once/layer.
// ---------------------------------------------------------------------------
__global__ void convt_kernel(const float* __restrict__ W, __bf16* __restrict__ Wt,
                             int K, int N) {
  int idx = blockIdx.x * blockDim.x + threadIdx.x;
  if (idx >= K * N) return;
  int n = idx / K, k = idx % K;
  Wt[idx] = (__bf16)W[(size_t)k * N + n];
}

// ---------------------------------------------------------------------------
// Token-shift mixes -> bf16 GEMM inputs: out_i = x + (x_prev - x) * mix_i.
// ---------------------------------------------------------------------------
__global__ void shift_mix6_kernel(const float* __restrict__ x,
    const float* __restrict__ mr, const float* __restrict__ mw,
    const float* __restrict__ mk, const float* __restrict__ mv,
    const float* __restrict__ ma, const float* __restrict__ mg,
    __bf16* __restrict__ xr, __bf16* __restrict__ xw, __bf16* __restrict__ xk,
    __bf16* __restrict__ xv, __bf16* __restrict__ xa, __bf16* __restrict__ xg) {
  int m = blockIdx.x;
  bool hp = (m % Tt) != 0;
  for (int c = threadIdx.x; c < Cdim; c += blockDim.x) {
    size_t idx = (size_t)m * Cdim + c;
    float cur  = x[idx];
    float prev = hp ? x[idx - Cdim] : 0.f;
    float xx = prev - cur;
    xr[idx] = (__bf16)(cur + xx * mr[c]);
    xw[idx] = (__bf16)(cur + xx * mw[c]);
    xk[idx] = (__bf16)(cur + xx * mk[c]);
    xv[idx] = (__bf16)(cur + xx * mv[c]);
    xa[idx] = (__bf16)(cur + xx * ma[c]);
    xg[idx] = (__bf16)(cur + xx * mg[c]);
  }
}

__global__ void shift_mix1_kernel(const float* __restrict__ x,
                                  const float* __restrict__ mk,
                                  __bf16* __restrict__ out) {
  int m = blockIdx.x;
  bool hp = (m % Tt) != 0;
  for (int c = threadIdx.x; c < Cdim; c += blockDim.x) {
    size_t idx = (size_t)m * Cdim + c;
    float cur  = x[idx];
    float prev = hp ? x[idx - Cdim] : 0.f;
    out[idx] = (__bf16)(cur + (prev - cur) * mk[c]);
  }
}

// ---------------------------------------------------------------------------
// WMMA GEMM: Y[M,N] (=|+=) epi(A[M,K] @ Wt[N,K]^T + bias[N]).
// A and Wt are bf16 in global memory. 128 threads = 4 waves; block tile 64x32;
// each wave: 16 rows x both 16-col fragments (A fragment reused, 2 WMMAs).
// LDS tiles stored in fragment order; staging is b128-only:
//   A(16x32): half=(K>>3)&1, e: K0 {0,8,16,24} -> e0 {0,0,8,8}, lane=(r&15)+16h
//   B(32x16): same K mapping as A,                              lane=n+16h
// ---------------------------------------------------------------------------
template <int EPI, bool ACC, bool OUTBF>
__global__ void gemm_wmma(const __bf16* __restrict__ A, const __bf16* __restrict__ Wt,
                          const float* __restrict__ bias, void* __restrict__ Yp,
                          int M, int N, int K) {
  (void)M;
  __shared__ v16bf AsF[128];   // 4 row-tiles (waves) x 32 lanes
  __shared__ v16bf BsF[64];    // 2 col-tiles x 32 lanes
  __bf16* As = (__bf16*)AsF;
  __bf16* Bs = (__bf16*)BsF;
  const int tid  = threadIdx.x;      // 0..127
  const int wave = tid >> 5;
  const int lane = tid & 31;
  const int half = lane >> 4;
  const int lm   = lane & 15;
  const int bm   = blockIdx.x;       // 64-row tile
  const int bn   = blockIdx.y;       // 32-col tile

  // Loop-invariant B staging mapping: thread -> (col 0..31, K-octet 0..3).
  const int  nB  = tid & 31;
  const int  K0B = (tid >> 5) * 8;             // {0,8,16,24}
  const int  hB  = (K0B >> 3) & 1;
  __bf16*    bdst = Bs + (((nB >> 4) * 32) + (nB & 15) + 16 * hB) * 16 + ((K0B & 16) >> 1);
  const __bf16* bsrc = &Wt[(size_t)(bn * 32 + nB) * K + K0B];

  v8f acc0 = {}, acc1 = {};
  for (int kt = 0; kt < K; kt += 32) {
    // ---- Stage A tile 64x32: one b128 load -> one b128 store per chunk ----
#pragma unroll
    for (int it = 0; it < 2; ++it) {
      int i  = tid + it * 128;       // 8-elem chunk 0..255
      int r  = i >> 2;               // row 0..63
      int c8 = (i & 3) * 8;          // K offset {0,8,16,24}
      uint4 d = *(const uint4*)&A[(size_t)(bm * 64 + r) * K + kt + c8];
      int h  = (c8 >> 3) & 1;
      int e0 = (c8 & 16) >> 1;       // {0,0,8,8}
      *(uint4*)(As + (((r >> 4) * 32 + (r & 15) + 16 * h) * 16 + e0)) = d;
    }
    // ---- Stage B tile 32x32 from transposed weights: b128 in, b128 out ----
    *(uint4*)bdst = *(const uint4*)(bsrc + kt);
    if (kt + 32 < K)
      __builtin_prefetch(&A[(size_t)(bm * 64 + (tid >> 1)) * K + kt + 32], 0, 0);
    __syncthreads();

    v16bf af  = AsF[wave * 32 + lane];
    v16bf bf0 = BsF[lane];
    v16bf bf1 = BsF[32 + lane];
    acc0 = __builtin_amdgcn_wmma_f32_16x16x32_bf16(false, af, false, bf0,
                                                   (short)0, acc0, false, false);
    acc1 = __builtin_amdgcn_wmma_f32_16x16x32_bf16(false, af, false, bf1,
                                                   (short)0, acc1, false, false);
    __syncthreads();
  }

  // C/D layout: VGPR j -> row j (+8 on upper half-lanes), col = lane&15.
  float*  Yf = (float*)Yp;
  __bf16* Yb = (__bf16*)Yp;
#pragma unroll
  for (int j = 0; j < 8; ++j) {
    int mrow = bm * 64 + wave * 16 + j + half * 8;
#pragma unroll
    for (int t = 0; t < 2; ++t) {
      int ncol = bn * 32 + t * 16 + lm;
      float v = (t == 0) ? acc0[j] : acc1[j];
      if (bias) v += bias[ncol];
      if (EPI == EPI_SIGMOID)      v = 1.f / (1.f + expf(-v));
      else if (EPI == EPI_TANH)    v = tanhf(v);
      else if (EPI == EPI_RELUSQ)  v = (v > 0.f) ? v * v : 0.f;
      size_t o = (size_t)mrow * N + ncol;
      if (OUTBF)    Yb[o] = (__bf16)v;
      else if (ACC) Yf[o] += v;
      else          Yf[o] = v;
    }
  }
}

// ---------------------------------------------------------------------------
// Decay: w = -softplus(-(w0 + wl)) - 0.5 ; wh = exp(-exp(w)).
// ---------------------------------------------------------------------------
__global__ void decay_kernel(const float* __restrict__ wl, const float* __restrict__ w0,
                             float* __restrict__ wh, int total) {
  int i = blockIdx.x * blockDim.x + threadIdx.x;
  if (i >= total) return;
  int c = i % Cdim;
  float u  = w0[c] + wl[i];
  float sp = (u > 0.f) ? log1pf(expf(-u)) : (-u + log1pf(expf(u)));  // softplus(-u)
  float w  = -sp - 0.5f;
  wh[i] = expf(-expf(w));
}

// v = v + (v_first - v) * sig
__global__ void vmix_kernel(float* __restrict__ v, const float* __restrict__ vf,
                            const float* __restrict__ sig, int total) {
  int i = blockIdx.x * blockDim.x + threadIdx.x;
  if (i >= total) return;
  v[i] = v[i] + (vf[i] - v[i]) * sig[i];
}

// ---------------------------------------------------------------------------
// Per-token prep: kk = normalize_head(k * k_k); km = k * (1 + (a-1)*k_a).
// ---------------------------------------------------------------------------
__global__ void prep_kernel(const float* __restrict__ k, const float* __restrict__ a,
                            const float* __restrict__ k_k, const float* __restrict__ k_a,
                            float* __restrict__ kkbuf, float* __restrict__ kmbuf) {
  int m = blockIdx.x;
  int tid = threadIdx.x;
  __shared__ float hss[Hh];
  if (tid < Hh) hss[tid] = 0.f;
  __syncthreads();
  float kkv[3];
#pragma unroll
  for (int i = 0; i < 3; ++i) {
    int c = tid * 3 + i;
    size_t idx = (size_t)m * Cdim + c;
    float kv = k[idx];
    float t  = kv * k_k[c];
    kkv[i] = t;
    atomicAdd(&hss[c / Nn], t * t);
    float av = a[idx];
    kmbuf[idx] = kv * (1.f + (av - 1.f) * k_a[c]);
  }
  __syncthreads();
#pragma unroll
  for (int i = 0; i < 3; ++i) {
    int c = tid * 3 + i;
    float nrm = fmaxf(sqrtf(hss[c / Nn]), 1e-12f);
    kkbuf[(size_t)m * Cdim + c] = kkv[i] / nrm;
  }
}

// ---------------------------------------------------------------------------
// Matrix-state recurrence. One block per (b,h); thread i owns state row i.
// S = S*diag(w) + (S@a) b^T + v k^T ; y = S@r ; with a=-kk, b=kk*a_sig.
// ---------------------------------------------------------------------------
__global__ void rwkv_recurrence_kernel(
    const float* __restrict__ wh, const float* __restrict__ km,
    const float* __restrict__ v,  const float* __restrict__ kk,
    const float* __restrict__ as, const float* __restrict__ r,
    float* __restrict__ y) {
  int bh = blockIdx.x;             // 0..B*H-1
  int b = bh / Hh, h = bh % Hh;
  int i = threadIdx.x;             // row 0..63
  __shared__ float wv[Nn], kv[Nn], vv[Nn], kkv[Nn], av[Nn], rv[Nn];
  float S[Nn];
#pragma unroll
  for (int j = 0; j < Nn; ++j) S[j] = 0.f;

  for (int t = 0; t < Tt; ++t) {
    size_t base = ((size_t)(b * Tt + t)) * Cdim + h * Nn;
    wv[i]  = wh[base + i];
    kv[i]  = km[base + i];
    vv[i]  = v [base + i];
    kkv[i] = kk[base + i];
    av[i]  = as[base + i];
    rv[i]  = r [base + i];
    __syncthreads();

    float sa = 0.f;
#pragma unroll
    for (int j = 0; j < Nn; ++j) sa -= S[j] * kkv[j];   // a_t = -kk
    float vi = vv[i];
    float yo = 0.f;
#pragma unroll
    for (int j = 0; j < Nn; ++j) {
      S[j] = S[j] * wv[j] + sa * (kkv[j] * av[j]) + vi * kv[j];
      yo  += S[j] * rv[j];
    }
    y[base + i] = yo;
    __syncthreads();
  }
}

// ---------------------------------------------------------------------------
// GroupNorm(eps=64e-5) per head + r_k bonus + output gate -> bf16 og.
// og = (GN(y)*lnx_w + lnx_b + dot(r*km*r_k)_h * v) * g
// ---------------------------------------------------------------------------
__global__ void gn_rk_gate_kernel(
    const float* __restrict__ y, const float* __restrict__ r,
    const float* __restrict__ km, const float* __restrict__ v,
    const float* __restrict__ g, const float* __restrict__ r_k,
    const float* __restrict__ lnw, const float* __restrict__ lnb,
    __bf16* __restrict__ og) {
  int m = blockIdx.x;
  int tid = threadIdx.x;
  __shared__ float hsum[Hh], hsq[Hh], hdot[Hh];
  if (tid < Hh) { hsum[tid] = 0.f; hsq[tid] = 0.f; hdot[tid] = 0.f; }
  __syncthreads();
  float yv[3];
#pragma unroll
  for (int i = 0; i < 3; ++i) {
    int c = tid * 3 + i;
    int h = c / Nn;
    size_t idx = (size_t)m * Cdim + c;
    float t = y[idx];
    yv[i] = t;
    atomicAdd(&hsum[h], t);
    atomicAdd(&hsq[h], t * t);
    atomicAdd(&hdot[h], r[idx] * km[idx] * r_k[c]);
  }
  __syncthreads();
#pragma unroll
  for (int i = 0; i < 3; ++i) {
    int c = tid * 3 + i;
    int h = c / Nn;
    size_t idx = (size_t)m * Cdim + c;
    float mu  = hsum[h] * (1.f / Nn);
    float var = hsq[h] * (1.f / Nn) - mu * mu;
    float gn  = (yv[i] - mu) * rsqrtf(var + 64e-5f) * lnw[c] + lnb[c];
    og[idx] = (__bf16)((gn + hdot[h] * v[idx]) * g[idx]);
  }
}

// ---------------------------------------------------------------------------
// Host orchestration.  Param order per layer (35 tensors):
// 0 x_r 1 x_w 2 x_k 3 x_v 4 x_a 5 x_g 6 w0 7 w1 8 w2 9 a0 10 a1 11 a2
// 12 v0 13 v1 14 v2 15 g1 16 g2 17 k_k 18 k_a 19 r_k 20 Wr 21 Wk 22 Wv 23 Wo
// 24 lnx_w 25 lnx_b 26 xk_ffn 27 Wk_ffn 28 Wv_ffn 29 ln0_w 30 ln0_b
// 31 ln1_w 32 ln1_b 33 ln2_w 34 ln2_b
// ---------------------------------------------------------------------------
extern "C" void kernel_launch(void* const* d_in, const int* in_sizes, int n_in,
                              void* d_out, int out_size, void* d_ws, size_t ws_size,
                              hipStream_t stream) {
  (void)in_sizes; (void)n_in; (void)out_size; (void)ws_size;
  const float* x_in = (const float*)d_in[0];
  float* ws = (float*)d_ws;
  const size_t MC = (size_t)Mtok * Cdim;

  // ---- f32 slots ----
  float* xcur   = ws + 0 * MC;
  float* lnbuf  = ws + 1 * MC;
  float* rb     = ws + 2 * MC;
  float* kb     = ws + 3 * MC;
  float* vb     = ws + 4 * MC;
  float* scr    = ws + 5 * MC;   // w-lora out, then v-sigmoid
  float* ab     = ws + 6 * MC;
  float* gb     = ws + 7 * MC;
  float* vfirst = ws + 8 * MC;
  float* whb    = ws + 9 * MC;
  float* kkb    = ws + 10 * MC;
  float* kmb    = ws + 11 * MC;
  float* yb     = ws + 12 * MC;

  // ---- bf16 region ----
  __bf16* bws   = (__bf16*)(ws + 13 * MC);
  size_t  boff  = 0;
  auto balloc = [&](size_t elems) { __bf16* p = bws + boff; boff += elems; return p; };
  __bf16* bxr   = balloc(MC);
  __bf16* bxw   = balloc(MC);
  __bf16* bxk   = balloc(MC);
  __bf16* bxv   = balloc(MC);
  __bf16* bxa   = balloc(MC);
  __bf16* bxg   = balloc(MC);
  __bf16* bog   = balloc(MC);
  __bf16* bffnkx= balloc(MC);
  __bf16* bffnh = balloc((size_t)Mtok * FFNDIM);
  __bf16* blora = balloc((size_t)Mtok * 128);
  __bf16* bghid = balloc((size_t)Mtok * 128);
  // transposed bf16 weights (converted per layer)
  __bf16* wWr = balloc((size_t)Cdim * Cdim);
  __bf16* wWk = balloc((size_t)Cdim * Cdim);
  __bf16* wWv = balloc((size_t)Cdim * Cdim);
  __bf16* wWo = balloc((size_t)Cdim * Cdim);
  __bf16* ww1 = balloc((size_t)Cdim * 64);
  __bf16* ww2 = balloc((size_t)Cdim * 64);
  __bf16* wa1 = balloc((size_t)Cdim * 64);
  __bf16* wa2 = balloc((size_t)Cdim * 64);
  __bf16* wv1 = balloc((size_t)Cdim * 32);
  __bf16* wv2 = balloc((size_t)Cdim * 32);
  __bf16* wg1 = balloc((size_t)Cdim * 128);
  __bf16* wg2 = balloc((size_t)Cdim * 128);
  __bf16* wKf = balloc((size_t)Cdim * FFNDIM);
  __bf16* wVf = balloc((size_t)Cdim * FFNDIM);

  hipMemcpyAsync(xcur, x_in, MC * sizeof(float), hipMemcpyDeviceToDevice, stream);

  const dim3 blk128(128);
  const int tot = Mtok * Cdim;
  const int eblk = (tot + 255) / 256;
  auto cgrid = [](int k, int n) { return dim3((unsigned)((k * n + 255) / 256)); };

  for (int layer = 0; layer < 4; ++layer) {
    auto p = [&](int i) { return (const float*)d_in[1 + layer * 35 + i]; };

    // Convert+transpose this layer's weights to bf16 [N,K].
    convt_kernel<<<cgrid(Cdim, Cdim), 256, 0, stream>>>(p(20), wWr, Cdim, Cdim);
    convt_kernel<<<cgrid(Cdim, Cdim), 256, 0, stream>>>(p(21), wWk, Cdim, Cdim);
    convt_kernel<<<cgrid(Cdim, Cdim), 256, 0, stream>>>(p(22), wWv, Cdim, Cdim);
    convt_kernel<<<cgrid(Cdim, Cdim), 256, 0, stream>>>(p(23), wWo, Cdim, Cdim);
    convt_kernel<<<cgrid(Cdim, 64),   256, 0, stream>>>(p(7),  ww1, Cdim, 64);
    convt_kernel<<<cgrid(64, Cdim),   256, 0, stream>>>(p(8),  ww2, 64, Cdim);
    convt_kernel<<<cgrid(Cdim, 64),   256, 0, stream>>>(p(10), wa1, Cdim, 64);
    convt_kernel<<<cgrid(64, Cdim),   256, 0, stream>>>(p(11), wa2, 64, Cdim);
    convt_kernel<<<cgrid(Cdim, 32),   256, 0, stream>>>(p(13), wv1, Cdim, 32);
    convt_kernel<<<cgrid(32, Cdim),   256, 0, stream>>>(p(14), wv2, 32, Cdim);
    convt_kernel<<<cgrid(Cdim, 128),  256, 0, stream>>>(p(15), wg1, Cdim, 128);
    convt_kernel<<<cgrid(128, Cdim),  256, 0, stream>>>(p(16), wg2, 128, Cdim);
    convt_kernel<<<cgrid(Cdim, FFNDIM), 256, 0, stream>>>(p(27), wKf, Cdim, FFNDIM);
    convt_kernel<<<cgrid(FFNDIM, Cdim), 256, 0, stream>>>(p(28), wVf, FFNDIM, Cdim);

    if (layer == 0)
      ln_kernel<<<Mtok, 256, 0, stream>>>(xcur, p(29), p(30), xcur);       // ln0
    ln_kernel<<<Mtok, 256, 0, stream>>>(xcur, p(31), p(32), lnbuf);        // ln1
    shift_mix6_kernel<<<Mtok, 256, 0, stream>>>(lnbuf, p(0), p(1), p(2), p(3), p(4), p(5),
                                                bxr, bxw, bxk, bxv, bxa, bxg);

    // r, k, v projections (f32 out, feed recurrence)
    gemm_wmma<EPI_NONE, false, false><<<dim3(Mtok / 64, Cdim / 32), blk128, 0, stream>>>(
        bxr, wWr, nullptr, rb, Mtok, Cdim, Cdim);
    gemm_wmma<EPI_NONE, false, false><<<dim3(Mtok / 64, Cdim / 32), blk128, 0, stream>>>(
        bxk, wWk, nullptr, kb, Mtok, Cdim, Cdim);
    gemm_wmma<EPI_NONE, false, false><<<dim3(Mtok / 64, Cdim / 32), blk128, 0, stream>>>(
        bxv, wWv, nullptr, vb, Mtok, Cdim, Cdim);

    // decay lora: tanh(xw @ w1) @ w2 -> decay
    gemm_wmma<EPI_TANH, false, true><<<dim3(Mtok / 64, 64 / 32), blk128, 0, stream>>>(
        bxw, ww1, nullptr, blora, Mtok, 64, Cdim);
    gemm_wmma<EPI_NONE, false, false><<<dim3(Mtok / 64, Cdim / 32), blk128, 0, stream>>>(
        blora, ww2, nullptr, scr, Mtok, Cdim, 64);
    decay_kernel<<<eblk, 256, 0, stream>>>(scr, p(6), whb, tot);

    // a (iclr): sigmoid(a0 + (xa @ a1) @ a2)
    gemm_wmma<EPI_NONE, false, true><<<dim3(Mtok / 64, 64 / 32), blk128, 0, stream>>>(
        bxa, wa1, nullptr, blora, Mtok, 64, Cdim);
    gemm_wmma<EPI_SIGMOID, false, false><<<dim3(Mtok / 64, Cdim / 32), blk128, 0, stream>>>(
        blora, wa2, p(9), ab, Mtok, Cdim, 64);

    // gate: sigmoid(xg @ g1) @ g2
    gemm_wmma<EPI_SIGMOID, false, true><<<dim3(Mtok / 64, 128 / 32), blk128, 0, stream>>>(
        bxg, wg1, nullptr, bghid, Mtok, 128, Cdim);
    gemm_wmma<EPI_NONE, false, false><<<dim3(Mtok / 64, Cdim / 32), blk128, 0, stream>>>(
        bghid, wg2, nullptr, gb, Mtok, Cdim, 128);

    // v residual mix (layers > 0)
    if (layer == 0) {
      hipMemcpyAsync(vfirst, vb, MC * sizeof(float), hipMemcpyDeviceToDevice, stream);
    } else {
      gemm_wmma<EPI_NONE, false, true><<<dim3(Mtok / 64, 32 / 32), blk128, 0, stream>>>(
          bxv, wv1, nullptr, blora, Mtok, 32, Cdim);
      gemm_wmma<EPI_SIGMOID, false, false><<<dim3(Mtok / 64, Cdim / 32), blk128, 0, stream>>>(
          blora, wv2, p(12), scr, Mtok, Cdim, 32);
      vmix_kernel<<<eblk, 256, 0, stream>>>(vb, vfirst, scr, tot);
    }

    prep_kernel<<<Mtok, 256, 0, stream>>>(kb, ab, p(17), p(18), kkb, kmb);
    rwkv_recurrence_kernel<<<Bb * Hh, Nn, 0, stream>>>(whb, kmb, vb, kkb, ab, rb, yb);
    gn_rk_gate_kernel<<<Mtok, 256, 0, stream>>>(yb, rb, kmb, vb, gb, p(19), p(24), p(25),
                                                bog);
    // xcur += og @ Wo
    gemm_wmma<EPI_NONE, true, false><<<dim3(Mtok / 64, Cdim / 32), blk128, 0, stream>>>(
        bog, wWo, nullptr, xcur, Mtok, Cdim, Cdim);

    // FFN (channel mix)
    ln_kernel<<<Mtok, 256, 0, stream>>>(xcur, p(33), p(34), lnbuf);        // ln2
    shift_mix1_kernel<<<Mtok, 256, 0, stream>>>(lnbuf, p(26), bffnkx);
    gemm_wmma<EPI_RELUSQ, false, true><<<dim3(Mtok / 64, FFNDIM / 32), blk128, 0, stream>>>(
        bffnkx, wKf, nullptr, bffnh, Mtok, FFNDIM, Cdim);
    gemm_wmma<EPI_NONE, true, false><<<dim3(Mtok / 64, Cdim / 32), blk128, 0, stream>>>(
        bffnh, wVf, nullptr, xcur, Mtok, Cdim, FFNDIM);
  }

  hipMemcpyAsync(d_out, xcur, MC * sizeof(float), hipMemcpyDeviceToDevice, stream);
}
